// practiceNet_27994596836177
// MI455X (gfx1250) — matrix operand order repaired and checked by reading, not compiled, not run on hardware
//
#include <hip/hip_runtime.h>
#include <hip/hip_bf16.h>
#include <math.h>

#define HH 96
#define WW 96
#define NPIX (HH * WW)   // 9216
#define BB 2
#define CD 64            // attention channel dim
#define SPLIT 4          // split-K factor over the key/value dimension
#define MSLICE (NPIX / SPLIT)        // 2304 keys per task (36 chunks of 64)
#define NBLK (NPIX / 32)             // 288 query blocks of 32 rows per batch
#define NTASK (BB * NBLK * SPLIT)    // 2304 wave-level partials
#define LROW 72                      // padded LDS row stride (bank-conflict-free)

typedef __attribute__((ext_vector_type(16))) _Float16     v16h;
typedef __attribute__((ext_vector_type(8)))  float        v8f;
typedef __attribute__((ext_vector_type(4)))  unsigned int v4u;

union FragH { v16h h; v4u q[2]; };

typedef decltype(__builtin_amdgcn_cvt_pkrtz(0.f, 0.f)) h2_t;
union PkU { h2_t h; int i; };

__device__ __forceinline__ v8f wmma_f16(v16h a, v16h b, v8f c) {
  // D = A(16x32 f16) * B(32x16 f16) + C(16x16 f32)
  return __builtin_amdgcn_wmma_f32_16x16x32_f16(false, a, false, b, (short)0, c,
                                                false, false);
}

// Fragment loads from a per-lane base pointer + compile-time element offset.
// A-layout (16x32): second 16B run at +8 elements. B-layout (32x16): at +8.
template <typename P>
__device__ __forceinline__ v16h loadA_ofs(P p, int ofs) {
  FragH f;
  f.q[0] = *(const v4u*)(p + ofs);
  f.q[1] = *(const v4u*)(p + ofs + 8);
  return f.h;
}
template <typename P>
__device__ __forceinline__ v16h loadB_ofs(P p, int ofs) {
  FragH f;
  f.q[0] = *(const v4u*)(p + ofs);
  f.q[1] = *(const v4u*)(p + ofs + 8);
  return f.h;
}

// ---------------- split-K flash attention, LDS-shared K/V chunks ----------------
// 4 waves per workgroup; each wave owns 32 query rows (128 per WG).  Every
// 64-key chunk is cooperatively staged once into LDS (double-buffered, one
// chunk ahead) and consumed by all four waves.  Row sums come from a P*ones
// WMMA; row max uses a packed-f16 xor-shuffle tree (the subtracted max cancels
// exactly in softmax, so f16 rounding is harmless).
__global__ __launch_bounds__(128) void attn_split(const _Float16* __restrict__ b16, // [B][C][N]
                                                  const _Float16* __restrict__ bT,  // [B][N][C]
                                                  float* __restrict__ pacc,         // [task][32][64]
                                                  float* __restrict__ pm,           // [task][32]
                                                  float* __restrict__ pl) {         // [task][32]
  __shared__ __align__(16) _Float16 Kbuf[2][64 * LROW]; // [m][c] rows of bT
  __shared__ __align__(16) _Float16 Vbuf[2][64 * LROW]; // [c][m] rows of b16
  __shared__ __align__(16) _Float16 Plds[4][32 * LROW]; // per-wave P tile

  const int tid  = threadIdx.x;
  const int wave = tid >> 5;
  const int lane = tid & 31;
  const int half = lane >> 4;
  const int r16  = lane & 15;

  const int wg     = blockIdx.x;
  const int split  = wg % SPLIT;
  const int wb     = wg / SPLIT;
  const int batch  = wb / (NPIX / 128);
  const int blk128 = wb % (NPIX / 128);
  const int n0     = blk128 * 128 + wave * 32;
  const int mbeg   = split * MSLICE;
  const int mend   = mbeg + MSLICE;
  const int task   = (batch * NBLK + blk128 * 4 + wave) * SPLIT + split;

  const _Float16* bQ = bT  + (size_t)batch * NPIX * CD; // [n][c], c-contiguous
  const _Float16* bV = b16 + (size_t)batch * CD * NPIX; // [c][n], n-contiguous

  // ---- cooperative chunk stage: 16KB per chunk across 128 threads ----
  auto loadChunk = [&](_Float16* dK, _Float16* dV, int m0) {
#pragma unroll
    for (int i = 0; i < 4; ++i) {
      int q = tid + i * 128;                 // 512 16B-chunks per tile
      int row = q >> 3, c8 = (q & 7) * 8;
      *(v4u*)(dK + row * LROW + c8) = *(const v4u*)(bQ + (size_t)(m0 + row) * CD + c8);
    }
#pragma unroll
    for (int i = 0; i < 4; ++i) {
      int q = tid + i * 128;
      int row = q >> 3, c8 = (q & 7) * 8;
      *(v4u*)(dV + row * LROW + c8) = *(const v4u*)(bV + (size_t)row * NPIX + m0 + c8);
    }
  };

  // Per-lane fragment base pointers (all fragment loads are immediate offsets).
  const _Float16* aBase = bQ + (size_t)(n0 + r16) * CD + (half ? 16 : 0);
  const _Float16* kB0 = (const _Float16*)&Kbuf[0][0] + r16 * LROW + (half ? 16 : 0);
  const _Float16* kB1 = (const _Float16*)&Kbuf[1][0] + r16 * LROW + (half ? 16 : 0);
  const _Float16* vB0 = (const _Float16*)&Vbuf[0][0] + r16 * LROW + (half ? 16 : 0);
  const _Float16* vB1 = (const _Float16*)&Vbuf[1][0] + r16 * LROW + (half ? 16 : 0);
  _Float16*       pSt = &Plds[wave][0];
  const _Float16* pB  = (const _Float16*)&Plds[wave][0] + r16 * LROW + (half ? 16 : 0);

  // All-ones B fragment for row-sum WMMAs.
  v16h ones;
  {
    FragH f;
    v4u o = {0x3C003C00u, 0x3C003C00u, 0x3C003C00u, 0x3C003C00u};
    f.q[0] = o; f.q[1] = o;
    ones = f.h;
  }

  // Query tiles: two row groups of 16, a[g*2+kt] (read once from global)
  v16h a[4];
#pragma unroll
  for (int g = 0; g < 2; ++g)
#pragma unroll
    for (int kt = 0; kt < 2; ++kt)
      a[g * 2 + kt] = loadA_ofs(aBase, g * 16 * CD + kt * 32);

  v8f acc[8];
#pragma unroll
  for (int t = 0; t < 8; ++t) { v8f z = {}; acc[t] = z; }
  float mrow[2][8], lsum[2][8];
#pragma unroll
  for (int g = 0; g < 2; ++g)
#pragma unroll
    for (int r = 0; r < 8; ++r) { mrow[g][r] = -1e30f; lsum[g][r] = 0.f; }

  // One 64-key chunk from LDS buffers kB/vB.
  auto compute = [&](const _Float16* kB, const _Float16* vB) {
    // ---- scores: per kt-half, 4 transient key fragments feed 8 WMMAs ----
    v8f S[2][4];
    {
      v16h kf[4];
#pragma unroll
      for (int mt = 0; mt < 4; ++mt) kf[mt] = loadB_ofs(kB, mt * 16 * LROW);
#pragma unroll
      for (int g = 0; g < 2; ++g)
#pragma unroll
        for (int mt = 0; mt < 4; ++mt) {
          v8f s = {};
          S[g][mt] = wmma_f16(a[g * 2 + 0], kf[mt], s);
        }
    }
    {
      v16h kf[4];
#pragma unroll
      for (int mt = 0; mt < 4; ++mt) kf[mt] = loadB_ofs(kB, mt * 16 * LROW + 32);
#pragma unroll
      for (int g = 0; g < 2; ++g)
#pragma unroll
        for (int mt = 0; mt < 4; ++mt)
          S[g][mt] = wmma_f16(a[g * 2 + 1], kf[mt], S[g][mt]);
    }

    // ---- online softmax per row group ----
#pragma unroll
    for (int g = 0; g < 2; ++g) {
      // chunk row-max, packed two rows per f16x2 register
      float cm8[8];
#pragma unroll
      for (int r = 0; r < 8; ++r)
        cm8[r] = fmaxf(fmaxf(S[g][0][r], S[g][1][r]), fmaxf(S[g][2][r], S[g][3][r]));
      PkU pk[4];
#pragma unroll
      for (int j = 0; j < 4; ++j)
        pk[j].h = __builtin_amdgcn_cvt_pkrtz(cm8[2 * j], cm8[2 * j + 1]);
#pragma unroll
      for (int st = 1; st <= 8; st <<= 1)
#pragma unroll
        for (int j = 0; j < 4; ++j) {
          PkU o; o.i = __shfl_xor(pk[j].i, st, 32);
          pk[j].h = __builtin_elementwise_max(pk[j].h, o.h);
        }
      // rescale running state with the (near-)max; the constant cancels exactly
#pragma unroll
      for (int r = 0; r < 8; ++r) {
        float cmf = (float)pk[r >> 1].h[r & 1];
        float mnew = fmaxf(mrow[g][r], cmf);
        float sc = __expf(mrow[g][r] - mnew);
        mrow[g][r] = mnew;
        lsum[g][r] *= sc;
        acc[g * 4 + 0][r] *= sc; acc[g * 4 + 1][r] *= sc;
        acc[g * 4 + 2][r] *= sc; acc[g * 4 + 3][r] *= sc;
      }
      // exponentiate straight into the LDS P tile (f16)
#pragma unroll
      for (int mt = 0; mt < 4; ++mt)
#pragma unroll
        for (int r = 0; r < 8; ++r)
          pSt[(g * 16 + r + 8 * half) * LROW + mt * 16 + r16] =
              (_Float16)__expf(S[g][mt][r] - mrow[g][r]);
    }

    // ---- reload P as A fragments (wave-private; in-order DS makes this safe) ----
    v16h pf[4];
#pragma unroll
    for (int g = 0; g < 2; ++g)
#pragma unroll
      for (int kt = 0; kt < 2; ++kt)
        pf[g * 2 + kt] = loadA_ofs(pB, g * 16 * LROW + kt * 32);

    // ---- row sums via P * ones (replaces a 64-bpermute reduction) ----
#pragma unroll
    for (int g = 0; g < 2; ++g) {
      v8f ls = {};
      ls = wmma_f16(pf[g * 2 + 0], ones, ls);
      ls = wmma_f16(pf[g * 2 + 1], ones, ls);
#pragma unroll
      for (int r = 0; r < 8; ++r) lsum[g][r] += ls[r];
    }

    // ---- out += P * V ----
    {
      v16h qf[4];
#pragma unroll
      for (int ct = 0; ct < 4; ++ct) qf[ct] = loadB_ofs(vB, ct * 16 * LROW);
#pragma unroll
      for (int g = 0; g < 2; ++g)
#pragma unroll
        for (int ct = 0; ct < 4; ++ct)
          acc[g * 4 + ct] = wmma_f16(pf[g * 2 + 0], qf[ct], acc[g * 4 + ct]);
    }
    {
      v16h qf[4];
#pragma unroll
      for (int ct = 0; ct < 4; ++ct) qf[ct] = loadB_ofs(vB, ct * 16 * LROW + 32);
#pragma unroll
      for (int g = 0; g < 2; ++g)
#pragma unroll
        for (int ct = 0; ct < 4; ++ct)
          acc[g * 4 + ct] = wmma_f16(pf[g * 2 + 1], qf[ct], acc[g * 4 + ct]);
    }
  };

  // ---- ping-pong over 36 chunks, staging one chunk ahead ----
  loadChunk(&Kbuf[0][0], &Vbuf[0][0], mbeg);
  __syncthreads();
  for (int m0 = mbeg; m0 < mend; m0 += 128) {
    loadChunk(&Kbuf[1][0], &Vbuf[1][0], m0 + 64);
    compute(kB0, vB0);
    __syncthreads();
    if (m0 + 128 < mend) loadChunk(&Kbuf[0][0], &Vbuf[0][0], m0 + 128);
    compute(kB1, vB1);
    __syncthreads();
  }

  // ---- write unnormalized partial + softmax stats for the combine pass ----
  float* pa = pacc + (size_t)task * 32 * 64;
#pragma unroll
  for (int g = 0; g < 2; ++g)
#pragma unroll
    for (int ct = 0; ct < 4; ++ct)
#pragma unroll
      for (int r = 0; r < 8; ++r) {
        int nl = g * 16 + r + 8 * half;
        int c  = ct * 16 + r16;
        pa[nl * 64 + c] = acc[g * 4 + ct][r];
      }
  if (r16 == 0) {
#pragma unroll
    for (int g = 0; g < 2; ++g)
#pragma unroll
      for (int r = 0; r < 8; ++r) {
        int nl = g * 16 + r + 8 * half;
        pm[task * 32 + nl] = mrow[g][r];
        pl[task * 32 + nl] = lsum[g][r];
      }
  }
}

// ---------------- merge the SPLIT partials with flash rescaling ----------------
__global__ void attn_combine(const float* __restrict__ pacc, const float* __restrict__ pm,
                             const float* __restrict__ pl, float* __restrict__ att) {
  int idx = blockIdx.x * blockDim.x + threadIdx.x;
  int total = BB * CD * NPIX;
  if (idx >= total) return;
  int n = idx % NPIX;
  int c = (idx / NPIX) % CD;
  int b = idx / (NPIX * CD);
  int nb = n / 32, nl = n % 32;
  int t0 = (b * NBLK + nb) * SPLIT;
  float M = -1e30f;
#pragma unroll
  for (int s = 0; s < SPLIT; ++s) M = fmaxf(M, pm[(t0 + s) * 32 + nl]);
  float l = 0.f, o = 0.f;
#pragma unroll
  for (int s = 0; s < SPLIT; ++s) {
    float w = __expf(pm[(t0 + s) * 32 + nl] - M);
    l += pl[(t0 + s) * 32 + nl] * w;
    o += pacc[(size_t)(t0 + s) * 2048 + nl * 64 + c] * w;
  }
  att[idx] = o / l;
}

// -------------------- direct conv + ReLU (tiny, L2-resident) --------------------
__global__ void conv2d_relu(const float* __restrict__ in, const float* __restrict__ w,
                            const float* __restrict__ bias, float* __restrict__ out,
                            int Cin, int Cout, int K, int pad) {
  int idx = blockIdx.x * blockDim.x + threadIdx.x;
  int total = BB * Cout * NPIX;
  if (idx >= total) return;
  int x  = idx % WW;
  int y  = (idx / WW) % HH;
  int oc = (idx / NPIX) % Cout;
  int b  = idx / (NPIX * Cout);
  float s = bias[oc];
  for (int ic = 0; ic < Cin; ++ic) {
    const float* inb = in + (size_t)(b * Cin + ic) * NPIX;
    const float* wk  = w  + (size_t)(oc * Cin + ic) * K * K;
    for (int ky = 0; ky < K; ++ky) {
      int yy = y + ky - pad;
      if ((unsigned)yy >= (unsigned)HH) continue;
      for (int kx = 0; kx < K; ++kx) {
        int xx = x + kx - pad;
        if ((unsigned)xx >= (unsigned)WW) continue;
        s += inb[yy * WW + xx] * wk[ky * K + kx];
      }
    }
  }
  out[idx] = fmaxf(s, 0.f);
}

// f32 -> f16 conversion producing both [c][n] and transposed [n][c] copies
__global__ void pack_f16(const float* __restrict__ x3, _Float16* __restrict__ b16,
                         _Float16* __restrict__ bT) {
  int idx = blockIdx.x * blockDim.x + threadIdx.x;
  int total = BB * CD * NPIX;
  if (idx >= total) return;
  int n = idx % NPIX;
  int c = (idx / NPIX) % CD;
  int b = idx / (NPIX * CD);
  _Float16 v = (_Float16)x3[idx];
  b16[idx] = v;
  bT[((size_t)b * NPIX + n) * CD + c] = v;
}

extern "C" void kernel_launch(void* const* d_in, const int* in_sizes, int n_in,
                              void* d_out, int out_size, void* d_ws, size_t ws_size,
                              hipStream_t stream) {
  (void)in_sizes; (void)n_in; (void)out_size; (void)ws_size;
  const float* x  = (const float*)d_in[0];
  const float* w1 = (const float*)d_in[1];  const float* b1 = (const float*)d_in[2];
  const float* w2 = (const float*)d_in[3];  const float* b2 = (const float*)d_in[4];
  const float* w3 = (const float*)d_in[5];  const float* b3 = (const float*)d_in[6];
  const float* w4 = (const float*)d_in[7];  const float* b4 = (const float*)d_in[8];
  const float* w5 = (const float*)d_in[9];  const float* b5 = (const float*)d_in[10];

  char* ws = (char*)d_ws;                       // ~34.4 MB scratch layout
  float*    x1   = (float*)   (ws + 0);         // [2][ 8][9216] f32
  float*    x2   = (float*)   (ws + 589824);    // [2][16][9216] f32
  float*    x3   = (float*)   (ws + 1769472);   // [2][64][9216] f32
  _Float16* b16  = (_Float16*)(ws + 6488064);   // [2][64][9216] f16
  _Float16* bT   = (_Float16*)(ws + 8847360);   // [2][9216][64] f16
  float*    att  = (float*)   (ws + 11206656);  // [2][64][9216] f32
  float*    x4   = (float*)   (ws + 15925248);  // [2][ 8][9216] f32
  float*    pm   = (float*)   (ws + 16515072);  // [2304][32] f32
  float*    pl   = (float*)   (ws + 16809984);  // [2304][32] f32
  float*    pacc = (float*)   (ws + 17104896);  // [2304][32][64] f32

  dim3 blk(256);
  conv2d_relu <<<dim3((BB *  8 * NPIX + 255) / 256), blk, 0, stream>>>(x,  w1, b1, x1, 1,  8,  3, 1);
  conv2d_relu <<<dim3((BB * 16 * NPIX + 255) / 256), blk, 0, stream>>>(x1, w2, b2, x2, 8,  16, 3, 1);
  conv2d_relu <<<dim3((BB * 64 * NPIX + 255) / 256), blk, 0, stream>>>(x2, w3, b3, x3, 16, 64, 5, 2);
  pack_f16    <<<dim3((BB * 64 * NPIX + 255) / 256), blk, 0, stream>>>(x3, b16, bT);
  attn_split  <<<dim3(BB * (NPIX / 128) * SPLIT), dim3(128), 0, stream>>>(b16, bT, pacc, pm, pl);
  attn_combine<<<dim3((BB * 64 * NPIX + 255) / 256), blk, 0, stream>>>(pacc, pm, pl, att);
  conv2d_relu <<<dim3((BB *  8 * NPIX + 255) / 256), blk, 0, stream>>>(att, w4, b4, x4, 64, 8, 3, 1);
  conv2d_relu <<<dim3((BB *  1 * NPIX + 255) / 256), blk, 0, stream>>>(x4, w5, b5, (float*)d_out, 8, 1, 3, 1);
}
